// Model_37134287241851
// MI455X (gfx1250) — matrix operand order repaired and checked by reading, not compiled, not run on hardware
//
#include <hip/hip_runtime.h>
#include <hip/hip_bf16.h>

typedef _Float16 half_t;
typedef __attribute__((ext_vector_type(16))) _Float16 v16h;
typedef __attribute__((ext_vector_type(8)))  _Float16 v8h;
typedef __attribute__((ext_vector_type(8)))  float    v8f;
typedef __attribute__((ext_vector_type(4)))  int      v4i;

#define GNN_N 32768
#define GNN_E 196608

// ---------------- flags / modes ----------------
enum { MODE_DENSE = 0, MODE_EDGE3 = 1, MODE_NODE2 = 2 };
enum {
  F_RELU  = 1,
  F_LN    = 2,
  F_RESID = 4,
  F_AGG   = 8,
  F_ST16  = 16,
  F_ST32  = 32
};

#define WSTRIDE 40  // LDS halves per weight column (32 + 8 pad: kills bank conflicts)

// ---- CDNA5 async global->LDS path (guarded so compile never breaks) ----
#if defined(__HIP_DEVICE_COMPILE__) && __has_builtin(__builtin_amdgcn_global_load_async_to_lds_b128)
#define HAVE_ASYNC_LDS 1
#else
#define HAVE_ASYNC_LDS 0
#endif

__device__ __forceinline__ void wait_async0() {
#if defined(__HIP_DEVICE_COMPILE__)
#if __has_builtin(__builtin_amdgcn_s_wait_asynccnt)
  __builtin_amdgcn_s_wait_asynccnt(0);
#else
  asm volatile("s_wait_asynccnt 0x0" ::: "memory");
#endif
#endif
}

// ============================================================================
// Fused WMMA GEMM:  out[M x 128] = act( A[M x K] * W[K x 128] + b )  (+LN/resid/agg)
// Block: 256 threads = 8 waves. Tile: 128 rows x 128 cols. Wave: 16 rows x 128.
// Wt is f16, pre-transposed: Wt[n*K + k]. Weight k-chunks double-buffered in LDS.
// ============================================================================
template <int MODE, int K, int FLAGS>
__global__ __launch_bounds__(256) void gemm128(
    const half_t* __restrict__ A0, const half_t* __restrict__ A1,
    const int* __restrict__ idxS, const int* __restrict__ idxR,
    const half_t* __restrict__ Wt, const float* __restrict__ bias,
    const float* __restrict__ ln_g, const float* __restrict__ ln_b,
    const float* __restrict__ resid, float* __restrict__ out32,
    half_t* __restrict__ out16, float* __restrict__ agg) {
  __shared__ half_t Wl[2][128 * WSTRIDE];

  const int tid     = threadIdx.x;
  const int wave    = tid >> 5;
  const int lane    = tid & 31;
  const int half_hi = lane >> 4;  // 0: lanes 0-15, 1: lanes 16-31
  const int l16     = lane & 15;
  const int row0    = blockIdx.x * 128 + wave * 16;
  const int am      = row0 + l16;  // row this lane supplies to A fragments

  int sidx = 0, ridx = 0;
  if (MODE == MODE_EDGE3) { sidx = idxS[am]; ridx = idxR[am]; }

  // cooperative weight-chunk staging: 256 threads x 16B cover 128 cols x 16 k (k 0-7, 16-23)
  const int wn   = tid >> 1;
  const int woff = (tid & 1) * 16;
  auto stage = [&](int kc, int buf) {
    const half_t* src = Wt + (size_t)wn * K + kc + woff;
    half_t* dst = &Wl[buf][wn * WSTRIDE + woff];
#if HAVE_ASYNC_LDS
    __builtin_amdgcn_global_load_async_to_lds_b128((v4i*)src, (v4i*)dst, 0, 0);
#else
    *(v8h*)dst = *(const v8h*)src;
#endif
  };

  v8f acc[8] = {};

  constexpr int NC = K / 32;
  stage(0, 0);
#if HAVE_ASYNC_LDS
  wait_async0();
#endif
  __syncthreads();

#pragma unroll
  for (int c = 0; c < NC; ++c) {
    const int kc  = c * 32;
    const int cur = c & 1;
    if (c + 1 < NC) stage(kc + 32, (c + 1) & 1);  // prefetch next chunk

    // ---- A fragment (ISA 16-bit A 16x32 layout: two contiguous v8h) ----
    const half_t* arow;
    if (MODE == MODE_DENSE) {
      arow = A0 + (size_t)am * K + kc;
    } else if (MODE == MODE_EDGE3) {  // concat(edge[e], node[s], node[r])
      if (kc < 128)      arow = A0 + (size_t)am * 128 + kc;
      else if (kc < 256) arow = A1 + (size_t)sidx * 128 + (kc - 128);
      else               arow = A1 + (size_t)ridx * 128 + (kc - 256);
    } else {  // MODE_NODE2: concat(node[i], agg[i])
      if (kc < 128) arow = A0 + (size_t)am * 128 + kc;
      else          arow = A1 + (size_t)am * 128 + (kc - 128);
    }
    v16h afrag;
    ((v8h*)&afrag)[0] = *(const v8h*)(arow + half_hi * 8);
    ((v8h*)&afrag)[1] = *(const v8h*)(arow + half_hi * 8 + 16);

    // ---- 8 n-tiles of B from LDS; accumulate ----
#pragma unroll
    for (int j = 0; j < 8; ++j) {
      const int n = j * 16 + l16;
      v16h bfrag;
      const v8h* bp = (const v8h*)(&Wl[cur][n * WSTRIDE + half_hi * 8]);
      ((v8h*)&bfrag)[0] = bp[0];
      ((v8h*)&bfrag)[1] = bp[2];
      acc[j] = __builtin_amdgcn_wmma_f32_16x16x32_f16(
          false, afrag, false, bfrag, (short)0, acc[j], false, false);
    }

    if (c + 1 < NC) {
#if HAVE_ASYNC_LDS
      wait_async0();
#endif
      __syncthreads();
    }
  }

  // ---------------- epilogue ----------------
  float gj[8], bbj[8];
#pragma unroll
  for (int j = 0; j < 8; ++j) {
    const int n = j * 16 + l16;
    const float bb = bias[n];
    if (FLAGS & F_LN) { gj[j] = ln_g[n]; bbj[j] = ln_b[n]; }
#pragma unroll
    for (int v = 0; v < 8; ++v) {
      float x = acc[j][v] + bb;
      if (FLAGS & F_RELU) x = fmaxf(x, 0.f);
      acc[j][v] = x;
    }
  }

  if (FLAGS & F_LN) {
#pragma unroll
    for (int v = 0; v < 8; ++v) {
      float s = 0.f, s2 = 0.f;
#pragma unroll
      for (int j = 0; j < 8; ++j) { const float x = acc[j][v]; s += x; s2 += x * x; }
#pragma unroll
      for (int m = 1; m < 16; m <<= 1) {
        s  += __shfl_xor(s, m, 32);
        s2 += __shfl_xor(s2, m, 32);
      }
      const float mu   = s * (1.f / 128.f);
      const float var  = s2 * (1.f / 128.f) - mu * mu;
      const float rstd = rsqrtf(var + 1e-5f);
#pragma unroll
      for (int j = 0; j < 8; ++j)
        acc[j][v] = (acc[j][v] - mu) * rstd * gj[j] + bbj[j];
    }
  }

#pragma unroll
  for (int v = 0; v < 8; ++v) {
    const int m = row0 + half_hi * 8 + v;
    int ar = 0;
    if (FLAGS & F_AGG) ar = idxR[m];
#pragma unroll
    for (int j = 0; j < 8; ++j) {
      const int n = j * 16 + l16;
      float x = acc[j][v];
      if (FLAGS & F_AGG)   atomicAdd(agg + (size_t)ar * 128 + n, x);  // segment_sum(e_new)
      if (FLAGS & F_RESID) x += resid[(size_t)m * 128 + n];
      if (FLAGS & F_ST32)  out32[(size_t)m * 128 + n] = x;
      if (FLAGS & F_ST16)  out16[(size_t)m * 128 + n] = (half_t)x;
    }
  }
}

// ============================================================================
// Helper kernels
// ============================================================================
__global__ void k_convert_wt(const float* __restrict__ W, half_t* __restrict__ Wt,
                             int K, int Kpad) {
  const int i = blockIdx.x * 256 + threadIdx.x;
  const int total = 128 * Kpad;
  if (i >= total) return;
  const int n = i / Kpad, k = i % Kpad;
  Wt[i] = (k < K) ? (half_t)W[(size_t)k * 128 + n] : (half_t)0.f;
}

__global__ void k_zero(float* p, int n) {
  const int i = blockIdx.x * 256 + threadIdx.x;
  if (i < n) p[i] = 0.f;
}

__global__ void k_f2h(const float* __restrict__ s, half_t* __restrict__ d, int n) {
  const int i = blockIdx.x * 256 + threadIdx.x;
  if (i < n) d[i] = (half_t)s[i];
}

__global__ void k_build_node_feat(const float* __restrict__ vel,
                                  const int* __restrict__ nt,
                                  const float* __restrict__ mean,
                                  const float* __restrict__ stdv,
                                  half_t* __restrict__ out) {
  const int i = blockIdx.x * 256 + threadIdx.x;
  if (i >= GNN_N) return;
  float f[11];
  f[0] = vel[2 * i];
  f[1] = vel[2 * i + 1];
#pragma unroll
  for (int k = 2; k < 11; ++k) f[k] = 0.f;
  const int t = nt[i];
  if (t >= 0 && t < 9) f[2 + t] = 1.f;
  half_t* o = out + (size_t)i * 32;
#pragma unroll
  for (int k = 0; k < 11; ++k) o[k] = (half_t)((f[k] - mean[k]) / stdv[k]);
#pragma unroll
  for (int k = 11; k < 32; ++k) o[k] = (half_t)0.f;
}

__global__ void k_build_edge_feat(const float* __restrict__ pos,
                                  const int* __restrict__ s,
                                  const int* __restrict__ r,
                                  const float* __restrict__ mean,
                                  const float* __restrict__ stdv,
                                  half_t* __restrict__ out) {
  const int e = blockIdx.x * 256 + threadIdx.x;
  if (e >= GNN_E) return;
  const int si = s[e], ri = r[e];
  const float rx = pos[2 * si] - pos[2 * ri];
  const float ry = pos[2 * si + 1] - pos[2 * ri + 1];
  const float nr = sqrtf(rx * rx + ry * ry);
  half_t* o = out + (size_t)e * 32;
  o[0] = (half_t)((rx - mean[0]) / stdv[0]);
  o[1] = (half_t)((ry - mean[1]) / stdv[1]);
  o[2] = (half_t)((nr - mean[2]) / stdv[2]);
#pragma unroll
  for (int k = 3; k < 32; ++k) o[k] = (half_t)0.f;
}

__global__ void k_decode(const half_t* __restrict__ h, const float* __restrict__ w,
                         const float* __restrict__ b, float* __restrict__ out) {
  const int i = blockIdx.x * 256 + threadIdx.x;
  if (i >= GNN_N) return;
  float a0 = b[0], a1 = b[1];
  const half_t* hr = h + (size_t)i * 128;
#pragma unroll 8
  for (int k = 0; k < 128; ++k) {
    const float x = (float)hr[k];
    a0 += x * w[2 * k];
    a1 += x * w[2 * k + 1];
  }
  out[2 * i]     = a0;
  out[2 * i + 1] = a1;
}

// ============================================================================
// Host orchestration
// ============================================================================
extern "C" void kernel_launch(void* const* d_in, const int* in_sizes, int n_in,
                              void* d_out, int out_size, void* d_ws, size_t ws_size,
                              hipStream_t stream) {
  (void)in_sizes; (void)n_in; (void)out_size; (void)ws_size;
  const int N = GNN_N, E = GNN_E;

  const float* velocity  = (const float*)d_in[0];
  const int*   node_type = (const int*)d_in[1];
  const float* mesh_pos  = (const float*)d_in[2];
  const int*   senders   = (const int*)d_in[3];
  const int*   receivers = (const int*)d_in[4];
  auto P = [&](int i) { return (const float*)d_in[i]; };
  // params leaf order (insertion order of the reference dict):
  // 5 node_mean, 6 node_std, 7 edge_mean, 8 edge_std,
  // 9..14 node_enc (w,b)x3, 15,16 node_enc_ln,
  // 17..22 edge_enc (w,b)x3, 23,24 edge_enc_ln,
  // 25+16t: step t: edge (w,b)x3, edge_ln(g,b), node (w,b)x3, node_ln(g,b)
  // 137..142: dec (w,b)x3

  // ---- workspace carve ----
  char* ws = (char*)d_ws;
  auto alloc = [&](size_t bytes) -> char* {
    char* r = ws;
    ws += (bytes + 255) & ~(size_t)255;
    return r;
  };
  half_t* node_feat16 = (half_t*)alloc((size_t)N * 32 * 2);
  half_t* edge_feat16 = (half_t*)alloc((size_t)E * 32 * 2);
  float*  node_lat    = (float*)alloc((size_t)N * 128 * 4);
  half_t* node_lat16  = (half_t*)alloc((size_t)N * 128 * 2);
  float*  edge_lat    = (float*)alloc((size_t)E * 128 * 4);
  half_t* edge_lat16  = (half_t*)alloc((size_t)E * 128 * 2);
  half_t* tmpA16      = (half_t*)alloc((size_t)E * 128 * 2);
  half_t* tmpB16      = (half_t*)alloc((size_t)E * 128 * 2);
  float*  agg         = (float*)alloc((size_t)N * 128 * 4);
  half_t* agg16       = (half_t*)alloc((size_t)N * 128 * 2);
  half_t* wpool       = (half_t*)alloc((size_t)128 * 8896 * 2);

  // ---- convert all 128-out weights to transposed f16 ----
  half_t* wp = wpool;
  auto conv = [&](const float* W, int K, int Kpad) -> half_t* {
    half_t* dst = wp;
    wp += (size_t)128 * Kpad;
    const int total = 128 * Kpad;
    k_convert_wt<<<(total + 255) / 256, 256, 0, stream>>>(W, dst, K, Kpad);
    return dst;
  };
  half_t* w_ne0 = conv(P(9), 11, 32);
  half_t* w_ne1 = conv(P(11), 128, 128);
  half_t* w_ne2 = conv(P(13), 128, 128);
  half_t* w_ee0 = conv(P(17), 3, 32);
  half_t* w_ee1 = conv(P(19), 128, 128);
  half_t* w_ee2 = conv(P(21), 128, 128);
  half_t* w_st[7][6];
  for (int t = 0; t < 7; ++t) {
    const int b = 25 + 16 * t;
    w_st[t][0] = conv(P(b + 0), 384, 384);
    w_st[t][1] = conv(P(b + 2), 128, 128);
    w_st[t][2] = conv(P(b + 4), 128, 128);
    w_st[t][3] = conv(P(b + 8), 256, 256);
    w_st[t][4] = conv(P(b + 10), 128, 128);
    w_st[t][5] = conv(P(b + 12), 128, 128);
  }
  half_t* w_d0 = conv(P(137), 128, 128);
  half_t* w_d1 = conv(P(139), 128, 128);

  // ---- build features ----
  k_build_node_feat<<<(N + 255) / 256, 256, 0, stream>>>(velocity, node_type, P(5), P(6), node_feat16);
  k_build_edge_feat<<<(E + 255) / 256, 256, 0, stream>>>(mesh_pos, senders, receivers, P(7), P(8), edge_feat16);

  const dim3 blk(256);
  const int gN = N / 128, gE = E / 128;

  // ---- node encoder ----
  gemm128<MODE_DENSE, 32, (F_RELU | F_ST16)><<<gN, blk, 0, stream>>>(
      node_feat16, nullptr, nullptr, nullptr, w_ne0, P(10), nullptr, nullptr,
      nullptr, nullptr, tmpA16, nullptr);
  gemm128<MODE_DENSE, 128, (F_RELU | F_ST16)><<<gN, blk, 0, stream>>>(
      tmpA16, nullptr, nullptr, nullptr, w_ne1, P(12), nullptr, nullptr,
      nullptr, nullptr, tmpB16, nullptr);
  gemm128<MODE_DENSE, 128, (F_LN | F_ST16 | F_ST32)><<<gN, blk, 0, stream>>>(
      tmpB16, nullptr, nullptr, nullptr, w_ne2, P(14), P(15), P(16),
      nullptr, node_lat, node_lat16, nullptr);

  // ---- edge encoder ----
  gemm128<MODE_DENSE, 32, (F_RELU | F_ST16)><<<gE, blk, 0, stream>>>(
      edge_feat16, nullptr, nullptr, nullptr, w_ee0, P(18), nullptr, nullptr,
      nullptr, nullptr, tmpA16, nullptr);
  gemm128<MODE_DENSE, 128, (F_RELU | F_ST16)><<<gE, blk, 0, stream>>>(
      tmpA16, nullptr, nullptr, nullptr, w_ee1, P(20), nullptr, nullptr,
      nullptr, nullptr, tmpB16, nullptr);
  gemm128<MODE_DENSE, 128, (F_LN | F_ST16 | F_ST32)><<<gE, blk, 0, stream>>>(
      tmpB16, nullptr, nullptr, nullptr, w_ee2, P(22), P(23), P(24),
      nullptr, edge_lat, edge_lat16, nullptr);

  // ---- 7 GraphNet blocks ----
  for (int t = 0; t < 7; ++t) {
    const int b = 25 + 16 * t;
    k_zero<<<(N * 128 + 255) / 256, 256, 0, stream>>>(agg, N * 128);

    // edge MLP: input = concat(edge_lat, node_lat[s], node_lat[r])  (gather fused)
    gemm128<MODE_EDGE3, 384, (F_RELU | F_ST16)><<<gE, blk, 0, stream>>>(
        edge_lat16, node_lat16, senders, receivers, w_st[t][0], P(b + 1),
        nullptr, nullptr, nullptr, nullptr, tmpA16, nullptr);
    gemm128<MODE_DENSE, 128, (F_RELU | F_ST16)><<<gE, blk, 0, stream>>>(
        tmpA16, nullptr, nullptr, nullptr, w_st[t][1], P(b + 3),
        nullptr, nullptr, nullptr, nullptr, tmpB16, nullptr);
    // L3: LN, scatter-add e_new into agg, residual into edge_lat (f32+f16)
    gemm128<MODE_DENSE, 128, (F_LN | F_AGG | F_RESID | F_ST16 | F_ST32)><<<gE, blk, 0, stream>>>(
        tmpB16, nullptr, nullptr, receivers, w_st[t][2], P(b + 5), P(b + 6), P(b + 7),
        edge_lat, edge_lat, edge_lat16, agg);

    k_f2h<<<(N * 128 + 255) / 256, 256, 0, stream>>>(agg, agg16, N * 128);

    // node MLP: input = concat(node_lat, agg)
    gemm128<MODE_NODE2, 256, (F_RELU | F_ST16)><<<gN, blk, 0, stream>>>(
        node_lat16, agg16, nullptr, nullptr, w_st[t][3], P(b + 9),
        nullptr, nullptr, nullptr, nullptr, tmpA16, nullptr);
    gemm128<MODE_DENSE, 128, (F_RELU | F_ST16)><<<gN, blk, 0, stream>>>(
        tmpA16, nullptr, nullptr, nullptr, w_st[t][4], P(b + 11),
        nullptr, nullptr, nullptr, nullptr, tmpB16, nullptr);
    gemm128<MODE_DENSE, 128, (F_LN | F_RESID | F_ST16 | F_ST32)><<<gN, blk, 0, stream>>>(
        tmpB16, nullptr, nullptr, nullptr, w_st[t][5], P(b + 13), P(b + 14), P(b + 15),
        node_lat, node_lat, node_lat16, nullptr);
  }

  // ---- decoder ----
  gemm128<MODE_DENSE, 128, (F_RELU | F_ST16)><<<gN, blk, 0, stream>>>(
      node_lat16, nullptr, nullptr, nullptr, w_d0, P(138), nullptr, nullptr,
      nullptr, nullptr, tmpA16, nullptr);
  gemm128<MODE_DENSE, 128, (F_RELU | F_ST16)><<<gN, blk, 0, stream>>>(
      tmpA16, nullptr, nullptr, nullptr, w_d1, P(140), nullptr, nullptr,
      nullptr, nullptr, tmpB16, nullptr);
  k_decode<<<(N + 255) / 256, 256, 0, stream>>>(tmpB16, P(141), P(142), (float*)d_out);
}
